// NNMultiheadAttention_62869731279490
// MI455X (gfx1250) — compile-verified, matrix operand order checked by reading, not run on hardware
//
#include <hip/hip_runtime.h>
#include <hip/hip_bf16.h>

// Problem constants (from reference): L=2048, N=2, E=1024, H=16, D=64
#define L_SEQ 2048
#define NB    2
#define EMB   1024
#define NH    16
#define HD    64
#define E3    3072
#define NTOK  4096   // L*N tokens

typedef __attribute__((ext_vector_type(16))) __bf16        v16bf;
typedef __attribute__((ext_vector_type(8)))  float         v8f;
typedef __attribute__((ext_vector_type(4)))  unsigned int  v4u;

// One WMMA operand fragment: 16 bf16 per lane = 8 VGPRs (32 bytes)
struct Frag {
  union {
    v16bf          bf;
    v4u            q[2];
    unsigned short us[16];
  };
};

static __device__ __forceinline__ unsigned short f2bf(float f) {
  union { float f; unsigned int u; } v; v.f = f;
  unsigned int u = v.u;
  u += 0x7fffu + ((u >> 16) & 1u);   // round-to-nearest-even
  return (unsigned short)(u >> 16);
}

// Load a 16x32 bf16 operand tile (A-layout, or B from an N-major source).
// Per CDNA5 ISA 16-bit A layout: lane l (half hh) holds row l,
// K elements {k0 + hh*8 + 0..7} and {k0 + 16 + hh*8 + 0..7} -> two b128 loads.
static __device__ __forceinline__ void load_frag(Frag& f, const unsigned short* base,
                                                 size_t ld, int l, int hh, int k0) {
  const unsigned short* p = base + (size_t)l * ld + (size_t)(k0 + hh * 8);
  f.q[0] = *(const v4u*)(p);
  f.q[1] = *(const v4u*)(p + 16);
}

static __device__ __forceinline__ v8f wmma_bf(const Frag& a, const Frag& b, v8f c) {
  // (neg_a, A, neg_b, B, c_mod, C, reuse_a, reuse_b)
  return __builtin_amdgcn_wmma_f32_16x16x32_bf16(false, a.bf, false, b.bf,
                                                 (short)0, c, false, false);
}

#define V8F_ZERO {0.f,0.f,0.f,0.f,0.f,0.f,0.f,0.f}

// ---------------------------------------------------------------------------
// Kernel 0: fp32 -> bf16 conversion of x, in_proj_weight, out_proj_weight
// ---------------------------------------------------------------------------
__global__ void convert_kernel(const float* __restrict__ x,
                               const float* __restrict__ wqkv,
                               const float* __restrict__ wout,
                               unsigned short* __restrict__ xbf,
                               unsigned short* __restrict__ wqkvbf,
                               unsigned short* __restrict__ woutbf) {
  size_t i = (size_t)blockIdx.x * blockDim.x + threadIdx.x;
  const size_t n0 = (size_t)NTOK * EMB;   // 4M
  const size_t n1 = (size_t)E3 * EMB;     // 3M
  const size_t n2 = (size_t)EMB * EMB;    // 1M
  if (i < n0)                { xbf[i] = f2bf(x[i]); }
  else if (i < n0 + n1)      { size_t j = i - n0;      wqkvbf[j] = f2bf(wqkv[j]); }
  else if (i < n0 + n1 + n2) { size_t j = i - n0 - n1; woutbf[j] = f2bf(wout[j]); }
}

// ---------------------------------------------------------------------------
// Software-pipelined 16x64 GEMM tile body shared by kernels 1 and 3.
// Double-buffered A (1 frag) + B (4 frags): next k-tile's loads are issued
// before the current k-tile's WMMAs so the matrix pipe overlaps L2 latency.
// ---------------------------------------------------------------------------
static __device__ __forceinline__ void gemm_tile_16x64(
    const unsigned short* __restrict__ Abase,   // tile row 0 of A (row-major, ld=EMB)
    const unsigned short* __restrict__ Bbase,   // tile col 0 of B^T (row-major, ld=EMB)
    int l, int hh, v8f acc[4]) {
  Frag a0, a1, b0[4], b1[4];

  load_frag(a0, Abase, EMB, l, hh, 0);
#pragma unroll
  for (int s = 0; s < 4; ++s)
    load_frag(b0[s], Bbase + (size_t)(s * 16) * EMB, EMB, l, hh, 0);

  for (int k0 = 0; k0 < EMB; k0 += 64) {
    // issue loads for k0+32 while computing k0
    load_frag(a1, Abase, EMB, l, hh, k0 + 32);
#pragma unroll
    for (int s = 0; s < 4; ++s)
      load_frag(b1[s], Bbase + (size_t)(s * 16) * EMB, EMB, l, hh, k0 + 32);
    __builtin_prefetch(Abase + (size_t)l * EMB + k0 + 64, 0, 3);
#pragma unroll
    for (int s = 0; s < 4; ++s)
      acc[s] = wmma_bf(a0, b0[s], acc[s]);

    // issue loads for k0+64 while computing k0+32
    if (k0 + 64 < EMB) {
      load_frag(a0, Abase, EMB, l, hh, k0 + 64);
#pragma unroll
      for (int s = 0; s < 4; ++s)
        load_frag(b0[s], Bbase + (size_t)(s * 16) * EMB, EMB, l, hh, k0 + 64);
    }
#pragma unroll
    for (int s = 0; s < 4; ++s)
      acc[s] = wmma_bf(a1, b1[s], acc[s]);
  }
}

// ---------------------------------------------------------------------------
// Kernel 1: QKV projection.  qkv[t][f] = sum_e x[t][e]*W[f][e] + b[f]
// One wave computes a 16x64 tile. Writes Q (scaled by 1/8) and K as
// per-head row-major [n][h][l][d], and V transposed as [n][h][d][l].
// ---------------------------------------------------------------------------
__global__ void __launch_bounds__(32)
qkv_gemm_kernel(const unsigned short* __restrict__ xbf,
                const unsigned short* __restrict__ wbf,
                const float* __restrict__ bias,
                unsigned short* __restrict__ Qb,
                unsigned short* __restrict__ Kb,
                unsigned short* __restrict__ Vt) {
  const int lane = threadIdx.x;
  const int l = lane & 15, hh = lane >> 4;
  const int row0 = blockIdx.y * 16;
  const int f0   = blockIdx.x * 64;

  v8f acc[4] = {V8F_ZERO, V8F_ZERO, V8F_ZERO, V8F_ZERO};
  gemm_tile_16x64(xbf + (size_t)row0 * EMB, wbf + (size_t)f0 * EMB, l, hh, acc);

#pragma unroll
  for (int s = 0; s < 4; ++s) {
#pragma unroll
    for (int r = 0; r < 8; ++r) {
      const int f = f0 + s * 16 + l;
      float val = acc[s][r] + bias[f];
      const int t  = row0 + 8 * hh + r;       // token = li*NB + n
      const int li = t >> 1, n = t & 1;
      const int c = f >> 10, e = f & 1023, head = e >> 6, d = e & 63;
      const size_t hoff = (size_t)(n * NH + head);
      if (c == 0)      Qb[(hoff * L_SEQ + li) * HD + d] = f2bf(val * 0.125f); // fold 1/sqrt(64)
      else if (c == 1) Kb[(hoff * L_SEQ + li) * HD + d] = f2bf(val);
      else             Vt[(hoff * HD + d) * L_SEQ + li] = f2bf(val);
    }
  }
}

// ---------------------------------------------------------------------------
// Kernel 2: flash attention. One wave owns a 16-row Q tile of one head and
// streams 32 keys/iter: 4 WMMAs for scores, online softmax in registers,
// P relaid out via LDS (C/D-layout -> A-layout), 4 WMMAs for P@V.
// K fragments are double-buffered across iterations; V^T fragments are
// loaded before the softmax so VALU work hides their latency.
// ---------------------------------------------------------------------------
__global__ void __launch_bounds__(32)
attn_kernel(const unsigned short* __restrict__ Qb,
            const unsigned short* __restrict__ Kb,
            const unsigned short* __restrict__ Vt,
            unsigned short* __restrict__ Obf) {
  __shared__ __align__(16) unsigned short Plds[16 * 32];
  const int lane = threadIdx.x;
  const int l = lane & 15, hh = lane >> 4;
  const int bid = blockIdx.x;
  const int qt = bid & 127;          // L/16 = 128 q-tiles
  const int h  = (bid >> 7) & 15;
  const int n  = bid >> 11;

  const size_t headLD = (size_t)(n * NH + h) * L_SEQ;
  const unsigned short* Qhead  = Qb + (headLD + (size_t)qt * 16) * HD;
  const unsigned short* Khead  = Kb + headLD * HD;
  const unsigned short* Vthead = Vt + (size_t)(n * NH + h) * HD * L_SEQ;

  Frag qa0, qa1;
  load_frag(qa0, Qhead, HD, l, hh, 0);
  load_frag(qa1, Qhead, HD, l, hh, 32);

  v8f o0 = V8F_ZERO, o1 = V8F_ZERO, o2 = V8F_ZERO, o3 = V8F_ZERO;
  float m[8], ls[8];
#pragma unroll
  for (int r = 0; r < 8; ++r) { m[r] = -1.0e30f; ls[r] = 0.f; }

  auto loadK = [&](Frag (&kb)[4], int j0) {
    load_frag(kb[0], Khead + (size_t)j0 * HD,        HD, l, hh, 0);
    load_frag(kb[1], Khead + (size_t)j0 * HD,        HD, l, hh, 32);
    load_frag(kb[2], Khead + (size_t)(j0 + 16) * HD, HD, l, hh, 0);
    load_frag(kb[3], Khead + (size_t)(j0 + 16) * HD, HD, l, hh, 32);
  };

  auto step = [&](Frag (&kbc)[4], Frag (&kbn)[4], int j0) {
    // --- scores S = (Q*scale) @ K^T for keys [j0, j0+32) ---
    v8f s1 = V8F_ZERO, s2 = V8F_ZERO;
    s1 = wmma_bf(qa0, kbc[0], s1);
    s1 = wmma_bf(qa1, kbc[1], s1);
    s2 = wmma_bf(qa0, kbc[2], s2);
    s2 = wmma_bf(qa1, kbc[3], s2);

    // issue next-iteration K loads and this iteration's V loads now; the
    // softmax VALU work below hides their latency.
    if (j0 + 32 < L_SEQ) loadK(kbn, j0 + 32);
    Frag vb0, vb1, vb2, vb3;
    load_frag(vb0, Vthead + j0,                      L_SEQ, l, hh, 0);
    load_frag(vb1, Vthead + (size_t)16 * L_SEQ + j0, L_SEQ, l, hh, 0);
    load_frag(vb2, Vthead + (size_t)32 * L_SEQ + j0, L_SEQ, l, hh, 0);
    load_frag(vb3, Vthead + (size_t)48 * L_SEQ + j0, L_SEQ, l, hh, 0);

    // --- online softmax: row r of half hh is matrix row 8*hh + r, columns
    //     live across the 16 lanes of the half -> shfl_xor {1,2,4,8} reduce ---
#pragma unroll
    for (int r = 0; r < 8; ++r) {
      float v = fmaxf(s1[r], s2[r]);
      v = fmaxf(v, __shfl_xor(v, 1, 32));
      v = fmaxf(v, __shfl_xor(v, 2, 32));
      v = fmaxf(v, __shfl_xor(v, 4, 32));
      v = fmaxf(v, __shfl_xor(v, 8, 32));
      const float mn   = fmaxf(m[r], v);
      const float corr = __expf(m[r] - mn);
      m[r] = mn;
      const float p1 = __expf(s1[r] - mn);
      const float p2 = __expf(s2[r] - mn);
      float rs = p1 + p2;
      rs += __shfl_xor(rs, 1, 32);
      rs += __shfl_xor(rs, 2, 32);
      rs += __shfl_xor(rs, 4, 32);
      rs += __shfl_xor(rs, 8, 32);
      ls[r] = ls[r] * corr + rs;
      o0[r] *= corr; o1[r] *= corr; o2[r] *= corr; o3[r] *= corr;
      // relayout P through LDS: row-major 16x32 bf16
      Plds[(8 * hh + r) * 32 + l]      = f2bf(p1);
      Plds[(8 * hh + r) * 32 + 16 + l] = f2bf(p2);
    }

    // --- O += P @ V ---
    Frag pa;
    load_frag(pa, Plds, 32, l, hh, 0);   // single wave: DS is in-order, no barrier
    o0 = wmma_bf(pa, vb0, o0);
    o1 = wmma_bf(pa, vb1, o1);
    o2 = wmma_bf(pa, vb2, o2);
    o3 = wmma_bf(pa, vb3, o3);
  };

  Frag kbA[4], kbB[4];
  loadK(kbA, 0);
  for (int j0 = 0; j0 < L_SEQ; j0 += 64) {
    step(kbA, kbB, j0);
    step(kbB, kbA, j0 + 32);
  }

  // --- normalize and scatter to token-major bf16 buffer (L,N,E) ---
#pragma unroll
  for (int r = 0; r < 8; ++r) {
    const float inv = 1.f / ls[r];
    const int qi = qt * 16 + 8 * hh + r;
    const size_t obase = ((size_t)qi * NB + n) * EMB + (size_t)h * HD;
    Obf[obase +  0 + l] = f2bf(o0[r] * inv);
    Obf[obase + 16 + l] = f2bf(o1[r] * inv);
    Obf[obase + 32 + l] = f2bf(o2[r] * inv);
    Obf[obase + 48 + l] = f2bf(o3[r] * inv);
  }
}

// ---------------------------------------------------------------------------
// Kernel 3: output projection. y[t][f] = sum_e O[t][e]*Wout[f][e] + b[f]
// fp32 store straight to d_out (L,N,E) == token-major.
// ---------------------------------------------------------------------------
__global__ void __launch_bounds__(32)
out_gemm_kernel(const unsigned short* __restrict__ Obf,
                const unsigned short* __restrict__ wbf,
                const float* __restrict__ bias,
                float* __restrict__ out) {
  const int lane = threadIdx.x;
  const int l = lane & 15, hh = lane >> 4;
  const int row0 = blockIdx.y * 16;
  const int f0   = blockIdx.x * 64;

  v8f acc[4] = {V8F_ZERO, V8F_ZERO, V8F_ZERO, V8F_ZERO};
  gemm_tile_16x64(Obf + (size_t)row0 * EMB, wbf + (size_t)f0 * EMB, l, hh, acc);

#pragma unroll
  for (int s = 0; s < 4; ++s) {
#pragma unroll
    for (int r = 0; r < 8; ++r) {
      const int f = f0 + s * 16 + l;
      const int t = row0 + 8 * hh + r;
      out[(size_t)t * EMB + f] = acc[s][r] + bias[f];
    }
  }
}

// ---------------------------------------------------------------------------
extern "C" void kernel_launch(void* const* d_in, const int* in_sizes, int n_in,
                              void* d_out, int out_size, void* d_ws, size_t ws_size,
                              hipStream_t stream) {
  (void)in_sizes; (void)n_in; (void)out_size; (void)ws_size;
  const float* x    = (const float*)d_in[0];
  const float* wqkv = (const float*)d_in[1];
  const float* bqkv = (const float*)d_in[2];
  const float* wout = (const float*)d_in[3];
  const float* bout = (const float*)d_in[4];
  float* out = (float*)d_out;

  char* ws = (char*)d_ws;                         // 48 MiB used, all 1 MiB aligned
  unsigned short* xbf    = (unsigned short*)(ws);                        //  8 MiB
  unsigned short* wqkvbf = (unsigned short*)(ws + ( 8ull << 20));        //  6 MiB
  unsigned short* woutbf = (unsigned short*)(ws + (14ull << 20));        //  2 MiB
  unsigned short* Qb     = (unsigned short*)(ws + (16ull << 20));        //  8 MiB
  unsigned short* Kb     = (unsigned short*)(ws + (24ull << 20));        //  8 MiB
  unsigned short* Vt     = (unsigned short*)(ws + (32ull << 20));        //  8 MiB
  unsigned short* Obf    = (unsigned short*)(ws + (40ull << 20));        //  8 MiB

  const size_t total = (size_t)NTOK * EMB + (size_t)E3 * EMB + (size_t)EMB * EMB;
  convert_kernel<<<(unsigned)((total + 255) / 256), 256, 0, stream>>>(
      x, wqkv, wout, xbf, wqkvbf, woutbf);

  qkv_gemm_kernel<<<dim3(E3 / 64, NTOK / 16), 32, 0, stream>>>(
      xbf, wqkvbf, bqkv, Qb, Kb, Vt);

  attn_kernel<<<NB * NH * (L_SEQ / 16), 32, 0, stream>>>(Qb, Kb, Vt, Obf);

  out_gemm_kernel<<<dim3(EMB / 64, NTOK / 16), 32, 0, stream>>>(
      Obf, woutbf, bout, out);
}
